// MagnitudeDiagnosticHRR_37082747633880
// MI455X (gfx1250) — compile-verified, matrix operand order checked by reading, not run on hardware
//
#include <hip/hip_runtime.h>
#include <hip/hip_bf16.h>
#include <cstdint>
#include <cstddef>

// ---------------------------------------------------------------------------
// MI455X (gfx1250) fused HRR block.
//   6 GEMMs (M=32768, K=N=1024) -> v_wmma_f32_16x16x32_bf16
//   A/B tiles staged to LDS via Tensor Data Mover (tensor_load_to_lds +
//   s_wait_tensorcnt) when the builtin is available, else reg staging.
//   chunked (64) complex cumsum + retrieve -> VALU/sincos
//   LayerNorm (wave32 shfl_xor reduction) + residual epilogue
// ---------------------------------------------------------------------------

typedef __attribute__((ext_vector_type(16))) __bf16 v16bf;
typedef __attribute__((ext_vector_type(8)))  float  v8f;
typedef __attribute__((ext_vector_type(4)))  unsigned int u32x4;
typedef __attribute__((ext_vector_type(8)))  int i32x8;
typedef __attribute__((ext_vector_type(4)))  int i32x4;

#define D_DIM   1024
#define M_ROWS  32768           // B*S = 4*8192
#define CHUNK   64
#define BM      128
#define BN      128
#define BK      32
#define NIT     (D_DIM / BK)    // 32 K-steps

#if defined(__HIP_DEVICE_COMPILE__) && \
    __has_builtin(__builtin_amdgcn_tensor_load_to_lds) && \
    __has_builtin(__builtin_amdgcn_s_wait_tensorcnt)
#define USE_TDM 1
#else
#define USE_TDM 0
#endif

// ---- bf16 helpers (round-to-nearest-even) ---------------------------------
__device__ __forceinline__ unsigned short f2bf(float x) {
  unsigned u = __float_as_uint(x);
  u += 0x7FFFu + ((u >> 16) & 1u);
  return (unsigned short)(u >> 16);
}
__device__ __forceinline__ float bf2f(unsigned short h) {
  return __uint_as_float(((unsigned)h) << 16);
}

union FragU { uint4 q[2]; v16bf v; };

#if USE_TDM
// ---------------------------------------------------------------------------
// TDM 2-D tile load: tile_rows x tile_k bf16 elements, row-major both sides.
// D# layout per CDNA5 ISA sec.8:
//   group0: [1:0]=count=1 | [63:32]=lds_addr | [120:64]=global tile addr |
//           [127:126]=type=2
//   group1: [17:16]=data_size(1 => 2 bytes) | tensor_dim0/1 | tile_dim0/1 |
//           tensor_dim0_stride (elements)
// 2-D tensor => descriptor groups 2/3 disabled (zeros).
// This toolchain exposes the 6-arg builtin:
//   (u32x4 g0, i32x8 g1, i32x4, i32x4, i32x8, i32 cpol)
// ---------------------------------------------------------------------------
__device__ __forceinline__ void tdm_load_2d_bf16(
    unsigned lds_addr, const unsigned short* gtile,
    unsigned tile_k, unsigned tile_rows,
    unsigned tensor_k, unsigned tensor_rows, unsigned row_stride_elems) {
  const unsigned long long ga = (unsigned long long)(uintptr_t)gtile;
  u32x4 g0 = { 1u,
               lds_addr,
               (unsigned)ga,
               (unsigned)((ga >> 32) & 0x1FFFFFFu) | (2u << 30) };
  i32x8 g1 = { (int)(1u << 16),                                   // data_size=2B
               (int)((tensor_k & 0xFFFFu) << 16),                 // td0 lo16
               (int)(((tensor_k >> 16) & 0xFFFFu) |
                     ((tensor_rows & 0xFFFFu) << 16)),            // td0 hi | td1 lo
               (int)(((tensor_rows >> 16) & 0xFFFFu) |
                     ((tile_k & 0xFFFFu) << 16)),                 // td1 hi | tile_dim0
               (int)(tile_rows & 0xFFFFu),                        // tile_dim1 | tile_dim2=0
               (int)row_stride_elems,                             // td0_stride lo32
               0,                                                 // td0_stride hi | td1_stride lo
               0 };
  i32x4 z4 = {0, 0, 0, 0};
  i32x8 z8 = {0, 0, 0, 0, 0, 0, 0, 0};
  __builtin_amdgcn_tensor_load_to_lds(g0, g1, z4, z4, z8, 0);
}
#endif

// ---------------------------------------------------------------------------
// x f32 -> bf16 pre-pass (removes conversion VALU from all GEMM inner loops)
// ---------------------------------------------------------------------------
__global__ __launch_bounds__(256) void xconv_k(const float* __restrict__ X,
                                               unsigned short* __restrict__ Xb) {
  const size_t i = ((size_t)blockIdx.x * 256 + threadIdx.x) * 4;
  const float4 f = *(const float4*)(X + i);
  const unsigned lo = (unsigned)f2bf(f.x) | ((unsigned)f2bf(f.y) << 16);
  const unsigned hi = (unsigned)f2bf(f.z) | ((unsigned)f2bf(f.w) << 16);
  *(uint2*)(Xb + i) = make_uint2(lo, hi);
}

// ---------------------------------------------------------------------------
// Weight convert + transpose: W[k][n] f32 -> Wt[n][k] bf16
// ---------------------------------------------------------------------------
__global__ __launch_bounds__(256) void wconv_k(const float* __restrict__ W,
                                               unsigned short* __restrict__ Wt) {
  const int idx = blockIdx.x * 256 + threadIdx.x;   // 0 .. 1M-1
  const int k = idx >> 10;
  const int n = idx & 1023;
  Wt[(size_t)n * D_DIM + k] = f2bf(W[(size_t)idx]);
}

// ---------------------------------------------------------------------------
// WMMA GEMM: C[M,N] = A[M,K](bf16, row stride lda) @ Wt^T (+bias) (+epilogue)
//   EPI: 0 = bf16 store, 1 = bf16 tanh(v)*aux[n], 2 = f32 v + aux[m*D+n]
// Fragment layout per ISA 7.12.2 (16-bit A 16x32 / B 32x16): lane L holds
// row L&15, K-base (L>>4)*8; VGPR pairs at K{kb..kb+7} and K{kb+16..kb+23}.
// ---------------------------------------------------------------------------
template<int EPI>
__global__ __launch_bounds__(256) void gemm_k(
    const unsigned short* __restrict__ A, int lda,
    const unsigned short* __restrict__ Wt,
    const float* __restrict__ bias, const float* __restrict__ aux,
    void* __restrict__ Cp) {
  __shared__ unsigned short lsA[2][BM * BK];   // 2 x 8KB
  __shared__ unsigned short lsB[2][BN * BK];   // 2 x 8KB

  const int t    = threadIdx.x;
  const int lane = t & 31;
  const int wid  = t >> 5;
  const int m0 = blockIdx.y * BM;
  const int n0 = blockIdx.x * BN;
  const int waveM = (wid & 3) * 32;   // 4 waves along M
  const int waveN = (wid >> 2) * 64;  // 2 waves along N

  v8f acc[2][4];
#pragma unroll
  for (int i = 0; i < 2; ++i)
#pragma unroll
    for (int j = 0; j < 4; ++j)
      acc[i][j] = v8f{0.f, 0.f, 0.f, 0.f, 0.f, 0.f, 0.f, 0.f};

  auto compute = [&](int buf) {
    const int r16 = lane & 15;
    const int kb  = (lane >> 4) * 8;
    FragU a[2], b[4];
#pragma unroll
    for (int mt = 0; mt < 2; ++mt) {
      const unsigned short* p = &lsA[buf][(waveM + mt * 16 + r16) * BK + kb];
      a[mt].q[0] = *(const uint4*)p;
      a[mt].q[1] = *(const uint4*)(p + 16);
    }
#pragma unroll
    for (int nt = 0; nt < 4; ++nt) {
      const unsigned short* p = &lsB[buf][(waveN + nt * 16 + r16) * BK + kb];
      b[nt].q[0] = *(const uint4*)p;
      b[nt].q[1] = *(const uint4*)(p + 16);
    }
#pragma unroll
    for (int mt = 0; mt < 2; ++mt)
#pragma unroll
      for (int nt = 0; nt < 4; ++nt)
        acc[mt][nt] = __builtin_amdgcn_wmma_f32_16x16x32_bf16(
            false, a[mt].v, false, b[nt].v, (short)0, acc[mt][nt], false, false);
  };

#if USE_TDM
  // ---- Tensor Data Mover staging: wave 0 drives the DMA ------------------
  const unsigned ldsA[2] = {(unsigned)(uintptr_t)&lsA[0][0],
                            (unsigned)(uintptr_t)&lsA[1][0]};
  const unsigned ldsB[2] = {(unsigned)(uintptr_t)&lsB[0][0],
                            (unsigned)(uintptr_t)&lsB[1][0]};
  auto issueTiles = [&](int buf, int kk) {
    tdm_load_2d_bf16(ldsA[buf], A + (size_t)m0 * lda + kk,
                     BK, BM, D_DIM, M_ROWS, (unsigned)lda);
    tdm_load_2d_bf16(ldsB[buf], Wt + (size_t)n0 * D_DIM + kk,
                     BK, BN, D_DIM, D_DIM, D_DIM);
  };
  if (wid == 0) {
    issueTiles(0, 0);
    __builtin_amdgcn_s_wait_tensorcnt(0);
  }
  __syncthreads();
  for (int it = 0; it < NIT; ++it) {
    const int buf = it & 1;
    if (wid == 0 && it + 1 < NIT) issueTiles(buf ^ 1, (it + 1) * BK);
    compute(buf);
    if (wid == 0 && it + 1 < NIT) __builtin_amdgcn_s_wait_tensorcnt(0);
    __syncthreads();
  }
#else
  // ---- Fallback: register-staged double buffer ---------------------------
  const int srow  = t >> 1;           // 0..127
  const int shalf = (t & 1) * 16;     // 0 or 16
  uint4 aregH[2], bregH[2];
  auto loadTile = [&](int kk) {
    const uint4* A4 = (const uint4*)(A + (size_t)(m0 + srow) * lda + kk + shalf);
    aregH[0] = A4[0]; aregH[1] = A4[1];
    const uint4* B4 = (const uint4*)(Wt + (size_t)(n0 + srow) * D_DIM + kk + shalf);
    bregH[0] = B4[0]; bregH[1] = B4[1];
  };
  auto stageLDS = [&](int buf) {
    unsigned short* da = &lsA[buf][srow * BK + shalf];
    ((uint4*)da)[0] = aregH[0];
    ((uint4*)da)[1] = aregH[1];
    unsigned short* db = &lsB[buf][srow * BK + shalf];
    ((uint4*)db)[0] = bregH[0];
    ((uint4*)db)[1] = bregH[1];
  };
  loadTile(0);
  stageLDS(0);
  __syncthreads();
  for (int it = 0; it < NIT; ++it) {
    const int buf = it & 1;
    if (it + 1 < NIT) loadTile((it + 1) * BK);
    if (it + 2 < NIT) {   // gfx1250 global_prefetch_b8
      __builtin_prefetch((const char*)Wt + ((size_t)(n0 + srow) * D_DIM +
                         (it + 2) * BK) * 2, 0, 0);
      __builtin_prefetch((const char*)A + ((size_t)(m0 + srow) * lda +
                         (it + 2) * BK) * 2, 0, 0);
    }
    compute(buf);
    if (it + 1 < NIT) stageLDS(buf ^ 1);
    __syncthreads();
  }
#endif

  // Epilogue: C 16x16 f32 layout -> VGPR r: M=r (lanes 0-15), M=r+8 (16-31)
  const int r16 = lane & 15;
  const int rowAdd = (lane >> 4) * 8;
#pragma unroll
  for (int mt = 0; mt < 2; ++mt) {
#pragma unroll
    for (int nt = 0; nt < 4; ++nt) {
      const int n = n0 + waveN + nt * 16 + r16;
      const float bv = bias[n];
      float psv = 0.f;
      if (EPI == 1) psv = aux[n];
#pragma unroll
      for (int r = 0; r < 8; ++r) {
        const int m = m0 + waveM + mt * 16 + rowAdd + r;
        float v = acc[mt][nt][r] + bv;
        if (EPI == 1) v = tanhf(v) * psv;
        if (EPI == 2) {
          float* C = (float*)Cp;
          C[(size_t)m * D_DIM + n] = v + aux[(size_t)m * D_DIM + n];
        } else {
          unsigned short* C = (unsigned short*)Cp;
          C[(size_t)m * D_DIM + n] = f2bf(v);
        }
      }
    }
  }
}

// ---------------------------------------------------------------------------
// Chunked complex accumulation + retrieve.
// One WG per 64-row chunk; each thread owns 4 feature columns (256*4 = 1024)
// and runs the 64-step recurrence in registers.
// ---------------------------------------------------------------------------
__global__ __launch_bounds__(256) void chunk_k(
    const unsigned short* __restrict__ vbf, const unsigned short* __restrict__ kp,
    const unsigned short* __restrict__ qp, float* __restrict__ ret) {
  const size_t r0 = (size_t)blockIdx.x * CHUNK;
  const int d = threadIdx.x;
  float cr[4] = {0.f, 0.f, 0.f, 0.f};
  float ci[4] = {0.f, 0.f, 0.f, 0.f};
  for (int row = 0; row < CHUNK; ++row) {
    const size_t base = (r0 + row) * D_DIM;
#pragma unroll
    for (int j = 0; j < 4; ++j) {
      const size_t col = base + d + j * 256;
      const float v = bf2f(vbf[col]);
      float sp, cp;
      __sincosf(bf2f(kp[col]), &sp, &cp);
      cr[j] += v * cp;
      ci[j] += v * sp;
      float sq, cq;
      __sincosf(bf2f(qp[col]), &sq, &cq);
      ret[col] = (cr[j] * cq + ci[j] * sq) * 0.03125f;  // 1/sqrt(1024)
    }
  }
}

// ---------------------------------------------------------------------------
// LayerNorm over D per row: one wave32 per row (8 rows / 256-thread WG).
// ---------------------------------------------------------------------------
__global__ __launch_bounds__(256) void ln_k(
    const float* __restrict__ ret, const float* __restrict__ g,
    const float* __restrict__ b, unsigned short* __restrict__ normed) {
  const int wid  = threadIdx.x >> 5;
  const int lane = threadIdx.x & 31;
  const size_t row = (size_t)blockIdx.x * 8 + wid;
  const float* rp = ret + row * D_DIM;
  float vals[32];
  float s = 0.f, s2 = 0.f;
#pragma unroll
  for (int i = 0; i < 32; ++i) {
    const float v = rp[lane + i * 32];
    vals[i] = v;
    s += v;
    s2 += v * v;
  }
#pragma unroll
  for (int off = 16; off > 0; off >>= 1) {
    s  += __shfl_xor(s,  off, 32);
    s2 += __shfl_xor(s2, off, 32);
  }
  const float mu  = s * (1.f / 1024.f);
  const float var = s2 * (1.f / 1024.f) - mu * mu;
  const float rs  = rsqrtf(var + 1e-5f);
  unsigned short* np = normed + row * D_DIM;
#pragma unroll
  for (int i = 0; i < 32; ++i) {
    const int dc = lane + i * 32;
    np[dc] = f2bf((vals[i] - mu) * rs * g[dc] + b[dc]);
  }
}

// ---------------------------------------------------------------------------
// Host orchestration. Workspace layout (bytes), total ~396 MB:
//   [0,12M)    : 6 transposed bf16 weight matrices (2MB each)
//   [12M,76M)  : x_bf16
//   [76M,140M) : key_bf   \ reused later by `retrieved` f32 (76M..204M)
//   [140M,204M): query_bf /
//   [204M,268M): value_bf  (reused later by `normed` bf16)
//   [268M,332M): key_phase bf16
//   [332M,396M): query_phase bf16
// ---------------------------------------------------------------------------
extern "C" void kernel_launch(void* const* d_in, const int* in_sizes, int n_in,
                              void* d_out, int out_size, void* d_ws, size_t ws_size,
                              hipStream_t stream) {
  (void)in_sizes; (void)n_in; (void)out_size; (void)ws_size;
  const float* x   = (const float*)d_in[0];
  const float* Wk  = (const float*)d_in[1];
  const float* bk  = (const float*)d_in[2];
  const float* Wv  = (const float*)d_in[3];
  const float* bv  = (const float*)d_in[4];
  const float* Wq  = (const float*)d_in[5];
  const float* bq  = (const float*)d_in[6];
  const float* Wkp = (const float*)d_in[7];
  const float* bkp = (const float*)d_in[8];
  const float* Wqp = (const float*)d_in[9];
  const float* bqp = (const float*)d_in[10];
  const float* ps  = (const float*)d_in[11];
  const float* lng = (const float*)d_in[12];
  const float* lnb = (const float*)d_in[13];
  const float* Wo  = (const float*)d_in[14];
  const float* bo  = (const float*)d_in[15];
  float* out = (float*)d_out;
  char* ws = (char*)d_ws;

  const size_t MB = 1024ull * 1024ull;
  unsigned short* wt_k   = (unsigned short*)(ws + 0 * MB);
  unsigned short* wt_v   = (unsigned short*)(ws + 2 * MB);
  unsigned short* wt_q   = (unsigned short*)(ws + 4 * MB);
  unsigned short* wt_kp  = (unsigned short*)(ws + 6 * MB);
  unsigned short* wt_qp  = (unsigned short*)(ws + 8 * MB);
  unsigned short* wt_o   = (unsigned short*)(ws + 10 * MB);
  unsigned short* x_bf   = (unsigned short*)(ws + 12 * MB);
  unsigned short* key_bf = (unsigned short*)(ws + 76 * MB);
  unsigned short* qry_bf = (unsigned short*)(ws + 140 * MB);
  unsigned short* val_bf = (unsigned short*)(ws + 204 * MB);
  unsigned short* kp_bf  = (unsigned short*)(ws + 268 * MB);
  unsigned short* qp_bf  = (unsigned short*)(ws + 332 * MB);
  float*          retr   = (float*)(ws + 76 * MB);            // reuse key+query
  unsigned short* normed = (unsigned short*)(ws + 204 * MB);  // reuse value

  const dim3 bW(256);
  const dim3 gW(D_DIM * D_DIM / 256);
  wconv_k<<<gW, bW, 0, stream>>>(Wk,  wt_k);
  wconv_k<<<gW, bW, 0, stream>>>(Wv,  wt_v);
  wconv_k<<<gW, bW, 0, stream>>>(Wq,  wt_q);
  wconv_k<<<gW, bW, 0, stream>>>(Wkp, wt_kp);
  wconv_k<<<gW, bW, 0, stream>>>(Wqp, wt_qp);
  wconv_k<<<gW, bW, 0, stream>>>(Wo,  wt_o);
  xconv_k<<<dim3((size_t)M_ROWS * D_DIM / 1024), bW, 0, stream>>>(x, x_bf);

  const dim3 gg(D_DIM / BN, M_ROWS / BM);   // (8, 256)
  const dim3 bb(256);
  gemm_k<0><<<gg, bb, 0, stream>>>(x_bf, D_DIM, wt_k, bk, nullptr, key_bf);
  gemm_k<0><<<gg, bb, 0, stream>>>(x_bf, D_DIM, wt_v, bv, nullptr, val_bf);
  gemm_k<0><<<gg, bb, 0, stream>>>(x_bf, D_DIM, wt_q, bq, nullptr, qry_bf);
  gemm_k<1><<<gg, bb, 0, stream>>>(key_bf, D_DIM, wt_kp, bkp, ps, kp_bf);
  gemm_k<1><<<gg, bb, 0, stream>>>(qry_bf, D_DIM, wt_qp, bqp, ps, qp_bf);

  chunk_k<<<dim3(M_ROWS / CHUNK), bb, 0, stream>>>(val_bf, kp_bf, qp_bf, retr);
  ln_k<<<dim3(M_ROWS / 8), bb, 0, stream>>>(retr, lng, lnb, normed);

  gemm_k<2><<<gg, bb, 0, stream>>>(normed, D_DIM, wt_o, bo, x, out);
}
